// LSTM_67602785239089
// MI455X (gfx1250) — matrix operand order repaired and checked by reading, not compiled, not run on hardware
//
#include <hip/hip_runtime.h>

typedef __attribute__((ext_vector_type(16))) __bf16 v16bf;
typedef __attribute__((ext_vector_type(8)))  __bf16 v8bf;
typedef __attribute__((ext_vector_type(8)))  float  v8f;

#define T_STEPS 512
#define ISZ 64
#define HSZ 128
#define LDS_STRIDE 136   // padded row stride in bf16 elements (272 B) to spread banks

#if __has_builtin(__builtin_amdgcn_tanhf)
// gfx1250 native v_tanh_f32: 1-instruction tanh, 3-instruction sigmoid
__device__ __forceinline__ float fast_tanh(float x) {
    return __builtin_amdgcn_tanhf(x);
}
__device__ __forceinline__ float fast_sigmoid(float x) {
    return fmaf(0.5f, __builtin_amdgcn_tanhf(0.5f * x), 0.5f);
}
#else
__device__ __forceinline__ float fast_sigmoid(float x) {
    float e = __builtin_amdgcn_exp2f(-x * 1.4426950408889634f);
    return __builtin_amdgcn_rcpf(1.0f + e);
}
__device__ __forceinline__ float fast_tanh(float x) {
    float e = __builtin_amdgcn_exp2f(-x * 2.8853900817779268f);
    return __builtin_amdgcn_rcpf(1.0f + e) * 2.0f - 1.0f;
}
#endif

// ---- Prologue 1: weights fp32 -> bf16 (row-major, B-fragment friendly) ----
__global__ void lstm_cvt_weights(const float* __restrict__ wih_f,
                                 const float* __restrict__ whh_f,
                                 __bf16* __restrict__ wih_b,
                                 __bf16* __restrict__ whh_b) {
    int i = blockIdx.x * 256 + threadIdx.x;           // 0 .. 65535
    if (i < 512 * 64)  wih_b[i] = (__bf16)wih_f[i];
    if (i < 512 * 128) whh_b[i] = (__bf16)whh_f[i];
}

// ---- Prologue 2: x fp32 -> bf16, K order kept so an A-fragment is one contiguous
//      32B per-lane load. Per 32-K block, store order [K0..31] (identity).
__global__ void lstm_cvt_x(const float* __restrict__ xf, __bf16* __restrict__ xb) {
    int i = blockIdx.x * 256 + threadIdx.x;           // element index into [B*T*64]
    const int k    = i & 63;
    const int base = i & ~63;
    const int kk   = k & 31, blk = k >> 5;
    const int grp  = kk >> 3;                         // 0..3
    const int ngrp = (0x3210 >> (grp * 4)) & 0xf;     // identity: 0->0, 1->1, 2->2, 3->3
    const int pos  = blk * 32 + ngrp * 8 + (k & 7);
    xb[base + pos] = (__bf16)xf[i];
}

__global__ __launch_bounds__(256) void lstm_kernel(
    const __bf16* __restrict__ xs,    // [B, T, 64] bf16, K in natural order
    const float*  __restrict__ b_ih,  // [512]
    const float*  __restrict__ b_hh,  // [512]
    const __bf16* __restrict__ wih,   // [512, 64]  bf16
    const __bf16* __restrict__ whh,   // [512, 128] bf16
    float* __restrict__ out)          // [B, T, 128] fp32
{
    __shared__ __bf16 hbuf[2][16 * LDS_STRIDE];       // double-buffered h tile (bf16)

    const int lane = threadIdx.x & 31;
    const int wave = threadIdx.x >> 5;   // 0..7 -> owns 16 hidden columns
    const int r16  = lane & 15;          // A-row / B-col / C-col within 16x16 tile
    const int half = lane >> 4;          // lane half selects K sub-range
    const int b0   = blockIdx.x * 16;    // batch tile base
    const int n0   = wave * 16;          // hidden column base

    // ---- Preload all weight B-fragments into VGPRs (lane = N, 16 contiguous K/lane) ----
    v16bf wihB[4][2];
    v16bf whhB[4][4];
    #pragma unroll
    for (int g = 0; g < 4; ++g) {
        const int n = g * HSZ + n0 + r16;
        #pragma unroll
        for (int kt = 0; kt < 2; ++kt)
            wihB[g][kt] = *(const v16bf*)(wih + n * ISZ + kt * 32 + half * 16);
        #pragma unroll
        for (int kt = 0; kt < 4; ++kt)
            whhB[g][kt] = *(const v16bf*)(whh + n * HSZ + kt * 32 + half * 16);
    }

    // ---- Bias splat into C-fragment shape (bias depends only on column) ----
    v8f bias[4];
    #pragma unroll
    for (int g = 0; g < 4; ++g) {
        const int idx = g * HSZ + n0 + r16;
        const float bv = b_ih[idx] + b_hh[idx];
        #pragma unroll
        for (int i = 0; i < 8; ++i) bias[g][i] = bv;
    }

    // ---- h0 = 0 in LDS buffer 0; c kept in registers ----
    for (int i = threadIdx.x; i < 16 * LDS_STRIDE; i += 256) hbuf[0][i] = (__bf16)0.0f;
    v8f c;
    #pragma unroll
    for (int i = 0; i < 8; ++i) c[i] = 0.0f;
    __syncthreads();

    // per-lane streaming pointers
    const __bf16* xq  = xs + (size_t)(b0 + r16) * T_STEPS * ISZ + half * 16; // A rows: lane%16
    const int colg = n0 + r16;
    float* outp = out + (size_t)b0 * T_STEPS * HSZ + colg;   // + (row*T + t)*128 later

    for (int t = 0; t < T_STEPS; ++t) {
        const int cur = t & 1, nxt = cur ^ 1;

        // prefetch next timestep's x into near caches (speculative; OOB drops silently)
        __builtin_prefetch(xq + ISZ, 0, 3);

        // ---- x A-fragments: one contiguous 32B load per K-tile (pre-converted bf16) ----
        v16bf xA0 = *(const v16bf*)(xq);
        v16bf xA1 = *(const v16bf*)(xq + 32);

        // ---- h A-fragments from LDS (2x ds_load_b128 per K-tile) ----
        v16bf hA[4];
        #pragma unroll
        for (int kt = 0; kt < 4; ++kt) {
            const __bf16* hp = &hbuf[cur][r16 * LDS_STRIDE + kt * 32 + half * 16];
            v8bf lo = *(const v8bf*)hp;
            v8bf hi = *(const v8bf*)(hp + 8);
            v16bf f;
            #pragma unroll
            for (int i = 0; i < 8; ++i) { f[i] = lo[i]; f[8 + i] = hi[i]; }
            hA[kt] = f;
        }

        // ---- gates = bias + x@Wih^T + h@Whh^T  (24 WMMAs / wave / step) ----
        v8f acc[4];
        #pragma unroll
        for (int g = 0; g < 4; ++g) {
            v8f a = bias[g];
            a = __builtin_amdgcn_wmma_f32_16x16x32_bf16(false, xA0, false, wihB[g][0], (short)0, a, false, false);
            a = __builtin_amdgcn_wmma_f32_16x16x32_bf16(false, xA1, false, wihB[g][1], (short)0, a, false, false);
            #pragma unroll
            for (int kt = 0; kt < 4; ++kt)
                a = __builtin_amdgcn_wmma_f32_16x16x32_bf16(false, hA[kt], false, whhB[g][kt], (short)0, a, false, false);
            acc[g] = a;
        }

        // ---- activations + state update ----
        float hn[8];
        #pragma unroll
        for (int i = 0; i < 8; ++i) {
            const float iv = fast_sigmoid(acc[0][i]);
            const float fv = fast_sigmoid(acc[1][i]);
            const float gv = fast_tanh(acc[2][i]);
            const float ov = fast_sigmoid(acc[3][i]);
            const float cn = fv * c[i] + iv * gv;
            c[i] = cn;
            hn[i] = ov * fast_tanh(cn);
            hbuf[nxt][(i + 8 * half) * LDS_STRIDE + colg] = (__bf16)hn[i];
        }

#if __has_builtin(__builtin_amdgcn_s_barrier_signal) && __has_builtin(__builtin_amdgcn_s_barrier_wait)
        // split barrier: make LDS h-stores visible, signal, then overlap global
        // output stores with other waves reaching the barrier.
        __asm volatile("s_wait_dscnt 0x0" ::: "memory");
        __builtin_amdgcn_s_barrier_signal(-1);
        #pragma unroll
        for (int i = 0; i < 8; ++i)
            outp[((size_t)(i + 8 * half) * T_STEPS + t) * HSZ] = hn[i];
        __builtin_amdgcn_s_barrier_wait(-1);
        __asm volatile("" ::: "memory");
#else
        #pragma unroll
        for (int i = 0; i < 8; ++i)
            outp[((size_t)(i + 8 * half) * T_STEPS + t) * HSZ] = hn[i];
        __syncthreads();
#endif
        xq += ISZ;
    }
}

extern "C" void kernel_launch(void* const* d_in, const int* in_sizes, int n_in,
                              void* d_out, int out_size, void* d_ws, size_t ws_size,
                              hipStream_t stream) {
    (void)in_sizes; (void)n_in; (void)out_size; (void)ws_size;
    const float* x    = (const float*)d_in[0];
    const float* W_ih = (const float*)d_in[1];
    const float* W_hh = (const float*)d_in[2];
    const float* b_ih = (const float*)d_in[3];
    const float* b_hh = (const float*)d_in[4];

    __bf16* xs    = (__bf16*)d_ws;                         // 1024*512*64 bf16 = 64 MiB
    __bf16* wih_b = xs + (size_t)1024 * T_STEPS * ISZ;     // 512*64  bf16
    __bf16* whh_b = wih_b + 512 * 64;                      // 512*128 bf16

    lstm_cvt_weights<<<256, 256, 0, stream>>>(W_ih, W_hh, wih_b, whh_b);
    lstm_cvt_x<<<(1024 * T_STEPS * ISZ) / 256, 256, 0, stream>>>(x, xs);
    lstm_kernel<<<64, 256, 0, stream>>>(xs, b_ih, b_hh, wih_b, whh_b, (float*)d_out);
}
